// SudokuGNN_76081050681381
// MI455X (gfx1250) — matrix operand order repaired and checked by reading, not compiled, not run on hardware
//
#include <hip/hip_runtime.h>
#include <math.h>

// ---------------------------------------------------------------------------
// Sudoku GNN, fused single kernel: one workgroup per batch item.
// h (f16, padded stride), agg (f32, padded stride), W_m/W_u (f16, fragment-
// swizzled) resident in LDS for all 32 iterations. All five GEMM shapes run
// on v_wmma_f32_16x16x32_f16 with bank-conflict-free LDS fragment layouts.
// Edge gather folded into WMMA A-fragment loads; scatter via ds_add_f32.
// Branchless GELU (v_exp_f32 + v_rcp_f32).
// ---------------------------------------------------------------------------

typedef __attribute__((ext_vector_type(16))) _Float16 v16h;
typedef __attribute__((ext_vector_type(8)))  _Float16 v8h;
typedef __attribute__((ext_vector_type(4)))  _Float16 v4h;
typedef __attribute__((ext_vector_type(8)))  float    v8f;
typedef __attribute__((ext_vector_type(4)))  float    v4f;

#define GSZ   9
#define BATCH 256
#define DDIM  96
#define TWO_D 192
#define NCELL 81
#define NNODE 108
#define NPAD  112      // 7 * 16
#define NEDGE 486
#define EPAD  496      // 31 * 16
#define ITERS 32

#define HSTR  104      // h row stride (halves): 52 dwords -> conflict-free A loads
#define ASTR  100      // agg row stride (f32): full-period bank mapping
#define T1STR 200      // encoder t1 row stride (halves), same bytes as agg region

// ---- shared memory layout (bytes) ----
#define OFF_H    0                            // NPAD*HSTR f16     = 23296
#define OFF_AGG  (OFF_H   + NPAD*HSTR*2)      // NPAD*ASTR f32     = 44800
#define OFF_WGT  (OFF_AGG + NPAD*ASTR*4)      // wm+wu f16 swz     = 73728
#define OFF_WM   OFF_WGT
#define OFF_WU   (OFF_WM  + TWO_D*DDIM*2)
#define OFF_SRC  (OFF_WGT + 2*TWO_D*DDIM*2)   // EPAD i32
#define OFF_DST  (OFF_SRC + EPAD*4)
#define OFF_BM   (OFF_DST + EPAD*4)
#define OFF_BU   (OFF_BM  + DDIM*4)
#define OFF_G2   (OFF_BU  + DDIM*4)
#define OFF_B2   (OFF_G2  + DDIM*4)
#define SMEM_BYTES (OFF_B2 + DDIM*4)          // = 147328 bytes
static_assert(SMEM_BYTES <= 160 * 1024, "want 2 workgroups per WGP");

// Branchless tanh-approx GELU: tanh(u) = 1 - 2/(exp2(2*log2e*u)+1).
__device__ __forceinline__ float gelu_f(float x) {
    const float c0 = 0.7978845608028654f;   // sqrt(2/pi)
    const float c1 = 0.044715f;
    float u = c0 * (x + c1 * x * x * x);
    float e = __builtin_amdgcn_exp2f(u * 2.8853900817779268f);  // exp(2u)
    float th = 1.0f - 2.0f * __builtin_amdgcn_rcpf(e + 1.0f);
    return 0.5f * x * (1.0f + th);
}

// A fragment (16x32 f16): lane = M-row, elems 0..7 = K[koff..+7],
// 8..15 = K[koff+16..+23], koff = 8*(lane>>4).
__device__ __forceinline__ v16h load_a16(const _Float16* p, int koff) {
    v8h lo = *(const v8h*)(p + koff);
    v8h hi = *(const v8h*)(p + koff + 16);
    v16h a;
#pragma unroll
    for (int i = 0; i < 8; ++i) { a[i] = lo[i]; a[i + 8] = hi[i]; }
    return a;
}

__device__ __forceinline__ v16h load_a32(const float* p, int koff) {
    v4f a0 = *(const v4f*)(p + koff);
    v4f a1 = *(const v4f*)(p + koff + 4);
    v4f a2 = *(const v4f*)(p + koff + 16);
    v4f a3 = *(const v4f*)(p + koff + 20);
    v16h a;
#pragma unroll
    for (int i = 0; i < 4; ++i) {
        a[i]      = (_Float16)a0[i];
        a[i + 4]  = (_Float16)a1[i];
        a[i + 8]  = (_Float16)a2[i];
        a[i + 12] = (_Float16)a3[i];
    }
    return a;
}

// Swizzled B fragment: per (kt,nt) block of 512 halves; lane l's lo 8 halves
// at l*8, hi 8 halves at 256 + l*8. Lanes 0..15 cover all 64 banks exactly.
__device__ __forceinline__ v16h load_b_swz(const _Float16* wS, int block, int lane) {
    const _Float16* p = wS + (block << 9) + (lane << 3);
    v8h lo = *(const v8h*)p;
    v8h hi = *(const v8h*)(p + 256);
    v16h b;
#pragma unroll
    for (int i = 0; i < 8; ++i) { b[i] = lo[i]; b[i + 8] = hi[i]; }
    return b;
}

// Stage a K x N (row-major, f32) weight matrix into fragment-swizzled f16 LDS.
__device__ __forceinline__ void stage_w_swz(_Float16* dst, const float* Wg,
                                            int K, int N, int tid) {
    int NT = N >> 4;
    for (int i = tid; i < K * N; i += 256) {
        int k = i / N, n = i % N;
        int kt = k >> 5, kl = k & 31, nt = n >> 4, j = n & 15;
        int off = ((kt * NT + nt) << 9) + ((kl >> 4) << 8) + (((((kl >> 3) & 1) << 4) + j) << 3) + (kl & 7);
        dst[off] = (_Float16)Wg[i];
    }
}

// LayerNorm over a 96-wide row, f16 in -> f16 out (in-place safe).
__device__ __forceinline__ void ln_row_f16(const _Float16* rin, _Float16* rout,
                                           const float* g, const float* bt) {
    float mu = 0.f, ss = 0.f;
    const v8h* p = (const v8h*)rin;
#pragma unroll
    for (int c = 0; c < 12; ++c) {
        v8h x = p[c];
#pragma unroll
        for (int i = 0; i < 8; ++i) { float v = (float)x[i]; mu += v; ss += v * v; }
    }
    mu *= (1.0f / DDIM);
    float inv = rsqrtf(ss * (1.0f / DDIM) - mu * mu + 1e-5f);
#pragma unroll
    for (int c = 0; c < 12; ++c) {
        v8h x = p[c];
        v8h y;
#pragma unroll
        for (int i = 0; i < 8; ++i) {
            float v = (float)x[i];
            y[i] = (_Float16)((v - mu) * inv * g[c * 8 + i] + bt[c * 8 + i]);
        }
        *(v8h*)(rout + c * 8) = y;
    }
}

// LayerNorm over a 96-wide row, f32 in -> f16 out.
__device__ __forceinline__ void ln_row_f32(const float* rin, _Float16* rout,
                                           const float* g, const float* bt) {
    float mu = 0.f, ss = 0.f;
    const v4f* p = (const v4f*)rin;
#pragma unroll
    for (int c = 0; c < 24; ++c) {
        v4f x = p[c];
        mu += x[0] + x[1] + x[2] + x[3];
        ss += x[0] * x[0] + x[1] * x[1] + x[2] * x[2] + x[3] * x[3];
    }
    mu *= (1.0f / DDIM);
    float inv = rsqrtf(ss * (1.0f / DDIM) - mu * mu + 1e-5f);
#pragma unroll
    for (int c = 0; c < 24; ++c) {
        v4f x = p[c];
        v4h y;
#pragma unroll
        for (int i = 0; i < 4; ++i)
            y[i] = (_Float16)((x[i] - mu) * inv * g[c * 4 + i] + bt[c * 4 + i]);
        *(v4h*)(rout + c * 4) = y;
    }
}

extern "C" __global__ __launch_bounds__(256)
void sudoku_gnn_fused(const int* __restrict__ puzzle, const int* __restrict__ edge_index,
                      const float* __restrict__ We1_g, const float* __restrict__ be1_g,
                      const float* __restrict__ We2_g, const float* __restrict__ be2_g,
                      const float* __restrict__ ln1g_g, const float* __restrict__ ln1b_g,
                      const float* __restrict__ Wm_g,  const float* __restrict__ bm_g,
                      const float* __restrict__ Wu_g,  const float* __restrict__ bu_g,
                      const float* __restrict__ ln2g_g, const float* __restrict__ ln2b_g,
                      const float* __restrict__ ln3g_g, const float* __restrict__ ln3b_g,
                      const float* __restrict__ Wd1_g, const float* __restrict__ bd1_g,
                      const float* __restrict__ Wd2_g, const float* __restrict__ bd2_g,
                      const float* __restrict__ Wd3_g, const float* __restrict__ bd3_g,
                      float* __restrict__ out) {
    extern __shared__ __align__(16) char smem[];
    _Float16* hS   = (_Float16*)(smem + OFF_H);    // NPAD rows, stride HSTR
    float*    aggS = (float*)   (smem + OFF_AGG);  // NPAD rows, stride ASTR
    _Float16* wmS  = (_Float16*)(smem + OFF_WM);   // swz: We2 / Wm / Wd1 / Wd2
    _Float16* wuS  = (_Float16*)(smem + OFF_WU);   // swz: Wu ; later t1d f16
    int*      srcS = (int*)     (smem + OFF_SRC);
    int*      dstS = (int*)     (smem + OFF_DST);
    float*    bmS  = (float*)   (smem + OFF_BM);
    float*    buS  = (float*)   (smem + OFF_BU);
    float*    g2S  = (float*)   (smem + OFF_G2);
    float*    b2S  = (float*)   (smem + OFF_B2);

    const int tid  = threadIdx.x;
    const int wave = tid >> 5;
    const int lane = tid & 31;
    const int l15  = lane & 15;
    const int koff = (lane >> 4) * 8;   // A-frag K offset
    const int mhi  = (lane >> 4) * 8;   // C-frag M offset
    const int bidx = blockIdx.x;

    // =========== stage: edges, params, zero h, We2(swz), Wu(swz) ===========
    for (int i = tid; i < EPAD; i += 256) {
        srcS[i] = (i < NEDGE) ? edge_index[i]         : NNODE;  // pad -> dummy row
        dstS[i] = (i < NEDGE) ? edge_index[NEDGE + i] : NNODE;
    }
    for (int i = tid; i < DDIM; i += 256) {
        bmS[i] = bm_g[i]; buS[i] = bu_g[i]; g2S[i] = ln2g_g[i]; b2S[i] = ln2b_g[i];
    }
    for (int i = tid; i < NPAD * HSTR; i += 256) hS[i] = (_Float16)0.0f;
    stage_w_swz(wmS, We2_g, TWO_D, DDIM, tid);   // encoder layer-2 weights
    stage_w_swz(wuS, Wu_g,  TWO_D, DDIM, tid);   // update weights (persist)

    // ---- encoder layer 1: t1 = gelu(x @ We1 + b1), f16 in agg region ----
    _Float16* t1h = (_Float16*)aggS;             // 112 rows, stride T1STR
    for (int idx = tid; idx < NNODE * TWO_D; idx += 256) {
        int node = idx / TWO_D, i = idx % TWO_D;
        float f0, f1, f2, f3, f4;
        if (node < NCELL) {
            int v = puzzle[bidx * NCELL + node];
            f0 = (float)v * (1.0f / GSZ);
            f1 = (v > 0) ? 1.0f : 0.0f;
            f2 = (float)(node / GSZ) * (1.0f / (GSZ - 1));
            f3 = (float)(node % GSZ) * (1.0f / (GSZ - 1));
            f4 = 0.0f;
        } else { f0 = f1 = f2 = f3 = 0.0f; f4 = 1.0f; }
        float a = be1_g[i]
                + f0 * We1_g[0 * TWO_D + i] + f1 * We1_g[1 * TWO_D + i]
                + f2 * We1_g[2 * TWO_D + i] + f3 * We1_g[3 * TWO_D + i]
                + f4 * We1_g[4 * TWO_D + i];
        t1h[node * T1STR + i] = (_Float16)gelu_f(a);
    }
    for (int i = tid; i < (NPAD - NNODE) * TWO_D; i += 256)     // zero pad rows
        t1h[(NNODE + i / TWO_D) * T1STR + (i % TWO_D)] = (_Float16)0.0f;
    __syncthreads();

    // ---- encoder layer 2 (WMMA): h_pre = t1 @ We2 + b2, one M-tile/wave ----
    if (wave < 7) {
        int mrow = wave * 16 + l15;
        v16h aF[6];
#pragma unroll
        for (int kt = 0; kt < 6; ++kt)
            aF[kt] = load_a16(t1h + mrow * T1STR + kt * 32, koff);
#pragma unroll
        for (int nt = 0; nt < 6; ++nt) {
            v8f c = {0.f, 0.f, 0.f, 0.f, 0.f, 0.f, 0.f, 0.f};
#pragma unroll
            for (int kt = 0; kt < 6; ++kt) {
                v16h bb = load_b_swz(wmS, kt * 6 + nt, lane);
                c = __builtin_amdgcn_wmma_f32_16x16x32_f16(false, aF[kt], false, bb,
                                                           (short)0, c, false, false);
            }
            int n = nt * 16 + l15;
            float bs = be2_g[n];
#pragma unroll
            for (int v = 0; v < 8; ++v) {
                int m = wave * 16 + v + mhi;
                if (m < NNODE) hS[m * HSTR + n] = (_Float16)(c[v] + bs);
            }
        }
    }
    __syncthreads();
    // LN1 in place on h rows; meanwhile overwrite wm slot with the loop weights
    for (int r = tid; r < NNODE; r += 256)
        ln_row_f16(hS + r * HSTR, hS + r * HSTR, ln1g_g, ln1b_g);
    stage_w_swz(wmS, Wm_g, TWO_D, DDIM, tid);
    __syncthreads();

    // ================= 32 message-passing iterations (LDS-resident) ========
    for (int it = 0; it < ITERS; ++it) {
        for (int i = tid; i < NPAD * ASTR; i += 256) aggS[i] = 0.0f;
        __syncthreads();

        // message GEMM: m = gelu([h[src]|h[dst]] @ Wm + bm); scatter to agg[dst]
        for (int et = wave; et < 31; et += 8) {
            int erow = et * 16 + l15;
            int nsrc = srcS[erow], ndst = dstS[erow];
            v16h aF[6];
#pragma unroll
            for (int kt = 0; kt < 6; ++kt) {
                int node = (kt < 3) ? nsrc : ndst;
                int k0   = (kt < 3 ? kt : kt - 3) * 32;
                aF[kt] = load_a16(hS + node * HSTR + k0, koff);
            }
            int dre[8];
#pragma unroll
            for (int v = 0; v < 8; ++v) dre[v] = dstS[et * 16 + v + mhi];
#pragma unroll
            for (int nt = 0; nt < 6; ++nt) {
                v8f c = {0.f, 0.f, 0.f, 0.f, 0.f, 0.f, 0.f, 0.f};
#pragma unroll
                for (int kt = 0; kt < 6; ++kt) {
                    v16h bb = load_b_swz(wmS, kt * 6 + nt, lane);
                    c = __builtin_amdgcn_wmma_f32_16x16x32_f16(false, aF[kt], false, bb,
                                                               (short)0, c, false, false);
                }
                int n = nt * 16 + l15;
                float bs = bmS[n];
#pragma unroll
                for (int v = 0; v < 8; ++v)
                    atomicAdd(&aggS[dre[v] * ASTR + n], gelu_f(c[v] + bs));
            }
        }
        __syncthreads();

        // update GEMM: u = gelu([h|agg] @ Wu + bu); r = h + u held in regs
        float uacc[6][8];
        if (wave < 7) {                       // wave-uniform; one M-tile per wave
            int mrow = wave * 16 + l15;
            v16h aF[6];
#pragma unroll
            for (int kt = 0; kt < 6; ++kt)
                aF[kt] = (kt < 3) ? load_a16(hS   + mrow * HSTR + kt * 32, koff)
                                  : load_a32(aggS + mrow * ASTR + (kt - 3) * 32, koff);
#pragma unroll
            for (int nt = 0; nt < 6; ++nt) {
                v8f c = {0.f, 0.f, 0.f, 0.f, 0.f, 0.f, 0.f, 0.f};
#pragma unroll
                for (int kt = 0; kt < 6; ++kt) {
                    v16h bb = load_b_swz(wuS, kt * 6 + nt, lane);
                    c = __builtin_amdgcn_wmma_f32_16x16x32_f16(false, aF[kt], false, bb,
                                                               (short)0, c, false, false);
                }
                int n = nt * 16 + l15;
                float bs = buS[n];
#pragma unroll
                for (int v = 0; v < 8; ++v) {
                    int m = wave * 16 + v + mhi;
                    uacc[nt][v] = (float)hS[m * HSTR + n] + gelu_f(c[v] + bs);
                }
            }
        }
        __syncthreads();                      // all reads of h/agg complete
        if (wave < 7) {                       // spill residual into agg
#pragma unroll
            for (int nt = 0; nt < 6; ++nt) {
                int n = nt * 16 + l15;
#pragma unroll
                for (int v = 0; v < 8; ++v)
                    aggS[(wave * 16 + v + mhi) * ASTR + n] = uacc[nt][v];
            }
        }
        __syncthreads();
        for (int r = tid; r < NNODE; r += 256)        // LN2: agg -> h (f16)
            ln_row_f32(aggS + r * ASTR, hS + r * HSTR, g2S, b2S);
        __syncthreads();
    }

    // ======================= decoder on 81 cell nodes ======================
    // phase 1: cln = LN3(h[:81]) as f16 in agg; Wd1(swz) into wm slot
    _Float16* clnS = (_Float16*)aggS;                 // 96x96 f16, stride 96
    for (int r = tid; r < NCELL; r += 256)
        ln_row_f16(hS + r * HSTR, clnS + r * DDIM, ln3g_g, ln3b_g);
    for (int i = tid; i < (96 - NCELL) * DDIM; i += 256)
        clnS[NCELL * DDIM + i] = (_Float16)0.0f;      // zero pad rows 81..95
    stage_w_swz(wmS, Wd1_g, DDIM, TWO_D, tid);        // 96 x 192, NT = 12
    __syncthreads();

    // phase 2 (WMMA): t1d = gelu(cln @ Wd1 + bd1) -> f16 in wu slot (96x192)
    _Float16* t1dS = wuS;
    if (wave < 6) {                                   // one M-tile per wave
        int mrow = wave * 16 + l15;
        v16h aF[3];
#pragma unroll
        for (int kt = 0; kt < 3; ++kt)
            aF[kt] = load_a16(clnS + mrow * DDIM + kt * 32, koff);
#pragma unroll
        for (int nt = 0; nt < 12; ++nt) {
            v8f c = {0.f, 0.f, 0.f, 0.f, 0.f, 0.f, 0.f, 0.f};
#pragma unroll
            for (int kt = 0; kt < 3; ++kt) {
                v16h bb = load_b_swz(wmS, kt * 12 + nt, lane);
                c = __builtin_amdgcn_wmma_f32_16x16x32_f16(false, aF[kt], false, bb,
                                                           (short)0, c, false, false);
            }
            int n = nt * 16 + l15;
            float bs = bd1_g[n];
#pragma unroll
            for (int v = 0; v < 8; ++v) {
                int m = wave * 16 + v + mhi;
                if (m < NCELL) t1dS[m * TWO_D + n] = (_Float16)gelu_f(c[v] + bs);
            }
        }
    }
    for (int i = tid; i < (96 - NCELL) * TWO_D; i += 256)
        t1dS[NCELL * TWO_D + i] = (_Float16)0.0f;     // zero pad rows 81..95
    __syncthreads();
    stage_w_swz(wmS, Wd2_g, TWO_D, DDIM, tid);
    __syncthreads();

    // phase 3 (WMMA): t2d = gelu(t1d @ Wd2 + bd2) -> f32 in agg (81x96)
    float* t2dS = aggS;
    if (wave < 6) {
        int mrow = wave * 16 + l15;
        v16h aF[6];
#pragma unroll
        for (int kt = 0; kt < 6; ++kt)
            aF[kt] = load_a16(t1dS + mrow * TWO_D + kt * 32, koff);
#pragma unroll
        for (int nt = 0; nt < 6; ++nt) {
            v8f c = {0.f, 0.f, 0.f, 0.f, 0.f, 0.f, 0.f, 0.f};
#pragma unroll
            for (int kt = 0; kt < 6; ++kt) {
                v16h bb = load_b_swz(wmS, kt * 6 + nt, lane);
                c = __builtin_amdgcn_wmma_f32_16x16x32_f16(false, aF[kt], false, bb,
                                                           (short)0, c, false, false);
            }
            int n = nt * 16 + l15;
            float bs = bd2_g[n];
#pragma unroll
            for (int v = 0; v < 8; ++v) {
                int m = wave * 16 + v + mhi;
                if (m < NCELL) t2dS[m * DDIM + n] = gelu_f(c[v] + bs);
            }
        }
    }
    __syncthreads();

    // phase 4: logits = t2d @ Wd3 + bd3  (81x9, K=96 — scalar, tiny)
    for (int idx = tid; idx < NCELL * GSZ; idx += 256) {
        int node = idx / GSZ, g = idx % GSZ;
        float acc = bd3_g[g];
        for (int j = 0; j < DDIM; ++j) acc += t2dS[node * DDIM + j] * Wd3_g[j * GSZ + g];
        out[bidx * (NCELL * GSZ) + idx] = acc;
    }
}

extern "C" void kernel_launch(void* const* d_in, const int* in_sizes, int n_in,
                              void* d_out, int out_size, void* d_ws, size_t ws_size,
                              hipStream_t stream) {
    (void)in_sizes; (void)n_in; (void)d_ws; (void)ws_size; (void)out_size;
    hipFuncSetAttribute((const void*)sudoku_gnn_fused,
                        hipFuncAttributeMaxDynamicSharedMemorySize, SMEM_BYTES);
    sudoku_gnn_fused<<<dim3(BATCH), dim3(256), SMEM_BYTES, stream>>>(
        (const int*)d_in[0],  (const int*)d_in[1],
        (const float*)d_in[2],  (const float*)d_in[3],
        (const float*)d_in[4],  (const float*)d_in[5],
        (const float*)d_in[6],  (const float*)d_in[7],
        (const float*)d_in[8],  (const float*)d_in[9],
        (const float*)d_in[10], (const float*)d_in[11],
        (const float*)d_in[12], (const float*)d_in[13],
        (const float*)d_in[14], (const float*)d_in[15],
        (const float*)d_in[16], (const float*)d_in[17],
        (const float*)d_in[18], (const float*)d_in[19],
        (const float*)d_in[20], (const float*)d_in[21],
        (float*)d_out);
}
